// UltrametricAttention_8924942041528
// MI455X (gfx1250) — matrix operand order repaired and checked by reading, not compiled, not run on hardware
//
#include <hip/hip_runtime.h>
#include <hip/hip_bf16.h>

#define EMBED  128
#define NHEADS 4
#define HDIM   32
#define BATCH  2
#define SEQ    1024
#define SALL   (BATCH * SEQ)   // 2048 flattened rows

typedef __attribute__((ext_vector_type(16))) _Float16 v16h;
typedef __attribute__((ext_vector_type(8)))  _Float16 v8h;
typedef __attribute__((ext_vector_type(8)))  float    v8f;
typedef __attribute__((ext_vector_type(4)))  float    v4f;

// ---------------------------------------------------------------------------
// WMMA tile loaders (wave32, v_wmma_f32_16x16x32_f16 operand layouts)
// ---------------------------------------------------------------------------

// A-matrix 16x32 f16, row-major source, leading dim `ld` (halves).
// lane 0-15: row M=lane, halves 0-7 -> K=0..7,  halves 8-15 -> K=16..23
// lane 16-31: row M=lane-16, halves 0-7 -> K=8..15, halves 8-15 -> K=24..31
__device__ __forceinline__ v16h load_A16x32(const _Float16* __restrict__ base,
                                            int ld, int row0, int k0, int lane) {
  int r   = lane & 15;
  int klo = (lane & 16) ? 8 : 0;
  const _Float16* p = base + (size_t)(row0 + r) * ld + k0 + klo;
  v8h lo = *(const v8h*)(p);        // K = klo..klo+7
  v8h hi = *(const v8h*)(p + 16);   // K = 16+klo..16+klo+7
  return __builtin_shufflevector(lo, hi, 0,1,2,3,4,5,6,7,8,9,10,11,12,13,14,15);
}

// B-matrix 32x16 f16, with B[k][n] = src[col0+n][k] (src row-major, ld halves).
// lane 0-15: col N=lane, halves 0-7 -> K=0..7, halves 8-15 -> K=16..23 ; lanes 16-31 -> K=8..15, K=24..31
__device__ __forceinline__ v16h load_B32x16(const _Float16* __restrict__ base,
                                            int ld, int col0, int k0, int lane) {
  int n   = lane & 15;
  int klo = (lane & 16) ? 8 : 0;
  const _Float16* p = base + (size_t)(col0 + n) * ld + k0 + klo;
  v8h lo = *(const v8h*)(p);
  v8h hi = *(const v8h*)(p + 16);
  return __builtin_shufflevector(lo, hi, 0,1,2,3,4,5,6,7,8,9,10,11,12,13,14,15);
}

// ---------------------------------------------------------------------------
// f32 -> f16 conversion
// ---------------------------------------------------------------------------
__global__ void cvt_f32_f16(const float* __restrict__ in,
                            _Float16* __restrict__ out, int n) {
  int i = blockIdx.x * blockDim.x + threadIdx.x;
  if (i < n) out[i] = (_Float16)in[i];
}

// ---------------------------------------------------------------------------
// Fused QKV projection: y = x @ W.T + b  for W in {Wq,Wk,Wv}
// One wave per (16-row, 16-col, matrix) tile. Q,K stored f32 (B*S, E).
// V stored f16 transposed: Vt[b][e][s]  (so attention B-tiles load contiguous).
// ---------------------------------------------------------------------------
__global__ void __launch_bounds__(32)
qkv_gemm(const _Float16* __restrict__ xh,
         const _Float16* __restrict__ Wqh,
         const _Float16* __restrict__ Wkh,
         const _Float16* __restrict__ Wvh,
         const float* __restrict__ bq,
         const float* __restrict__ bk,
         const float* __restrict__ bv,
         float* __restrict__ Q,
         float* __restrict__ K,
         _Float16* __restrict__ Vt) {
  int lane = threadIdx.x;
  int row0 = blockIdx.x * 16;
  int nt   = blockIdx.y;            // 0..23
  int mat  = nt >> 3;               // 0=Q 1=K 2=V
  int n0   = (nt & 7) * 16;

  const _Float16* W    = (mat == 0) ? Wqh : (mat == 1) ? Wkh : Wvh;
  const float*    bias = (mat == 0) ? bq  : (mat == 1) ? bk  : bv;

  v8f acc = {};
#pragma unroll
  for (int k0 = 0; k0 < EMBED; k0 += 32) {
    v16h a = load_A16x32(xh, EMBED, row0, k0, lane);
    v16h b = load_B32x16(W,  EMBED, n0,   k0, lane);
    acc = __builtin_amdgcn_wmma_f32_16x16x32_f16(false, a, false, b,
                                                 (short)0, acc, false, false);
  }

  int n     = lane & 15;
  int rbase = (lane & 16) ? 8 : 0;
  float bb  = bias[n0 + n];

  if (mat == 2) {
#pragma unroll
    for (int j = 0; j < 8; ++j) {
      int r  = row0 + rbase + j;        // flattened b*SEQ+s
      int b_ = r >> 10;                 // / SEQ
      int s  = r & (SEQ - 1);
      Vt[((size_t)(b_ * EMBED + n0 + n)) * SEQ + s] = (_Float16)(acc[j] + bb);
    }
  } else {
    float* O = (mat == 0) ? Q : K;
#pragma unroll
    for (int j = 0; j < 8; ++j) {
      int r = row0 + rbase + j;
      O[(size_t)r * EMBED + n0 + n] = acc[j] + bb;
    }
  }
}

// ---------------------------------------------------------------------------
// Ultrametric (Chebyshev) attention, flash-style online softmax.
// One wave per (b, h, 16-row s-tile). K chunk staged in LDS (pitch 33 =>
// conflict-free), logits in f32 VALU, probabilities fed to WMMA as f16 A,
// V tiles loaded from transposed Vt as WMMA B. Output f16 (B,S,E).
// ---------------------------------------------------------------------------
__global__ void __launch_bounds__(32)
ultra_attn(const float* __restrict__ Q,
           const float* __restrict__ K,
           const _Float16* __restrict__ Vt,
           _Float16* __restrict__ Att) {
  __shared__ float Ks[32][33];
  __shared__ float rowBcast[16];
  __shared__ float rowL[16];

  int lane = threadIdx.x;
  int s0   = blockIdx.x * 16;
  int h    = blockIdx.y;
  int b    = blockIdx.z;

  int m     = lane & 15;
  int klo   = (lane & 16) ? 8 : 0;
  int rbase = (lane & 16) ? 8 : 0;

  // q row for this lane's M-row, kept in registers (32 f32)
  float q[HDIM];
  const float* qp = Q + ((size_t)(b * SEQ + s0 + m)) * EMBED + h * HDIM;
#pragma unroll
  for (int d4 = 0; d4 < HDIM / 4; ++d4) {
    v4f t = *(const v4f*)(qp + 4 * d4);
    q[4 * d4 + 0] = t[0]; q[4 * d4 + 1] = t[1];
    q[4 * d4 + 2] = t[2]; q[4 * d4 + 3] = t[3];
  }

  const float*    Kbase = K + ((size_t)b * SEQ) * EMBED + h * HDIM;
  const _Float16* Vb    = Vt + (size_t)b * EMBED * SEQ;

  float m_run = -INFINITY;
  float l_run = 0.f;
  v8f acc0 = {};
  v8f acc1 = {};

  for (int t0 = 0; t0 < SEQ; t0 += 32) {
    __syncthreads();   // previous iteration done with Ks / rowBcast
    // stage K chunk: lane loads row t0+lane (32 f32)
    const float* kp = Kbase + (size_t)(t0 + lane) * EMBED;
#pragma unroll
    for (int d4 = 0; d4 < HDIM / 4; ++d4) {
      v4f t = *(const v4f*)(kp + 4 * d4);
      Ks[lane][4 * d4 + 0] = t[0]; Ks[lane][4 * d4 + 1] = t[1];
      Ks[lane][4 * d4 + 2] = t[2]; Ks[lane][4 * d4 + 3] = t[3];
    }
    __syncthreads();

    if (t0 + 32 < SEQ)   // hint next chunk into cache
      __builtin_prefetch(Kbase + (size_t)(t0 + 32 + lane) * EMBED, 0, 1);

    // 16 logits per lane, laid out exactly as the WMMA A-tile
    float logit[16];
#pragma unroll
    for (int e = 0; e < 16; ++e) {
      int tt = (e < 8) ? (klo + e) : (16 + klo + (e - 8));
      float dist = 0.f;
#pragma unroll
      for (int d = 0; d < HDIM; ++d)
        dist = fmaxf(dist, fabsf(q[d] - Ks[tt][d]));
      logit[e] = -dist;
    }

    // online softmax: row stats live in lane pair (m, m+16)
    float cm = logit[0];
#pragma unroll
    for (int e = 1; e < 16; ++e) cm = fmaxf(cm, logit[e]);
    cm = fmaxf(cm, __shfl_xor(cm, 16, 32));
    float m_new = fmaxf(m_run, cm);
    float scale = __expf(m_run - m_new);
    m_run = m_new;

    float p[16];
    float rs = 0.f;
#pragma unroll
    for (int e = 0; e < 16; ++e) {
      p[e] = __expf(logit[e] - m_new);
      rs += p[e];
    }
    rs += __shfl_xor(rs, 16, 32);
    l_run = l_run * scale + rs;

    // broadcast per-row rescale factor to the C-layout lanes
    if (lane < 16) rowBcast[lane] = scale;
    __syncthreads();
#pragma unroll
    for (int j = 0; j < 8; ++j) {
      float sc = rowBcast[rbase + j];
      acc0[j] *= sc;
      acc1[j] *= sc;
    }

    // probabilities -> f16 A tile
    v16h a;
#pragma unroll
    for (int e = 0; e < 16; ++e) a[e] = (_Float16)p[e];

    // V tiles: B[k][n] = V[b][t0+k][h*32 + n (+16)]
    v16h vb0 = load_B32x16(Vb, SEQ, h * HDIM + 0,  t0, lane);
    v16h vb1 = load_B32x16(Vb, SEQ, h * HDIM + 16, t0, lane);

    acc0 = __builtin_amdgcn_wmma_f32_16x16x32_f16(false, a, false, vb0,
                                                  (short)0, acc0, false, false);
    acc1 = __builtin_amdgcn_wmma_f32_16x16x32_f16(false, a, false, vb1,
                                                  (short)0, acc1, false, false);
  }

  if (lane < 16) rowL[lane] = l_run;
  __syncthreads();

  int n = lane & 15;
#pragma unroll
  for (int j = 0; j < 8; ++j) {
    float inv = 1.0f / rowL[rbase + j];
    size_t o = ((size_t)(b * SEQ + s0 + rbase + j)) * EMBED + h * HDIM;
    Att[o + n]      = (_Float16)(acc0[j] * inv);
    Att[o + 16 + n] = (_Float16)(acc1[j] * inv);
  }
}

// ---------------------------------------------------------------------------
// Output projection: out = attended @ Wo.T + bo   (f32 out)
// ---------------------------------------------------------------------------
__global__ void __launch_bounds__(32)
out_proj(const _Float16* __restrict__ Att,
         const _Float16* __restrict__ Woh,
         const float* __restrict__ bo,
         float* __restrict__ out) {
  int lane = threadIdx.x;
  int row0 = blockIdx.x * 16;
  int n0   = blockIdx.y * 16;

  v8f acc = {};
#pragma unroll
  for (int k0 = 0; k0 < EMBED; k0 += 32) {
    v16h a = load_A16x32(Att, EMBED, row0, k0, lane);
    v16h b = load_B32x16(Woh, EMBED, n0,   k0, lane);
    acc = __builtin_amdgcn_wmma_f32_16x16x32_f16(false, a, false, b,
                                                 (short)0, acc, false, false);
  }

  int n     = lane & 15;
  int rbase = (lane & 16) ? 8 : 0;
  float bb  = bo[n0 + n];
#pragma unroll
  for (int j = 0; j < 8; ++j)
    out[(size_t)(row0 + rbase + j) * EMBED + n0 + n] = acc[j] + bb;
}

// ---------------------------------------------------------------------------
// Host launcher
// ---------------------------------------------------------------------------
extern "C" void kernel_launch(void* const* d_in, const int* in_sizes, int n_in,
                              void* d_out, int out_size, void* d_ws, size_t ws_size,
                              hipStream_t stream) {
  const float* x  = (const float*)d_in[0];
  const float* Wq = (const float*)d_in[1];
  const float* bq = (const float*)d_in[2];
  const float* Wk = (const float*)d_in[3];
  const float* bk = (const float*)d_in[4];
  const float* Wv = (const float*)d_in[5];
  const float* bv = (const float*)d_in[6];
  const float* Wo = (const float*)d_in[7];
  const float* bo = (const float*)d_in[8];
  float* out = (float*)d_out;

  // workspace carve-up (~3.8 MB total)
  char* w = (char*)d_ws;
  _Float16* xh  = (_Float16*)w; w += (size_t)SALL * EMBED * sizeof(_Float16);
  _Float16* Wqh = (_Float16*)w; w += (size_t)EMBED * EMBED * sizeof(_Float16);
  _Float16* Wkh = (_Float16*)w; w += (size_t)EMBED * EMBED * sizeof(_Float16);
  _Float16* Wvh = (_Float16*)w; w += (size_t)EMBED * EMBED * sizeof(_Float16);
  _Float16* Woh = (_Float16*)w; w += (size_t)EMBED * EMBED * sizeof(_Float16);
  float*    Qb  = (float*)w;    w += (size_t)SALL * EMBED * sizeof(float);
  float*    Kb  = (float*)w;    w += (size_t)SALL * EMBED * sizeof(float);
  _Float16* Vt  = (_Float16*)w; w += (size_t)SALL * EMBED * sizeof(_Float16);
  _Float16* Att = (_Float16*)w; w += (size_t)SALL * EMBED * sizeof(_Float16);
  (void)in_sizes; (void)n_in; (void)out_size; (void)ws_size;

  // 1) f16 conversions
  {
    int n = SALL * EMBED;
    cvt_f32_f16<<<(n + 255) / 256, 256, 0, stream>>>(x, xh, n);
    int nw = EMBED * EMBED;
    cvt_f32_f16<<<(nw + 255) / 256, 256, 0, stream>>>(Wq, Wqh, nw);
    cvt_f32_f16<<<(nw + 255) / 256, 256, 0, stream>>>(Wk, Wkh, nw);
    cvt_f32_f16<<<(nw + 255) / 256, 256, 0, stream>>>(Wv, Wvh, nw);
    cvt_f32_f16<<<(nw + 255) / 256, 256, 0, stream>>>(Wo, Woh, nw);
  }

  // 2) fused QKV projection (WMMA)
  qkv_gemm<<<dim3(SALL / 16, 24), 32, 0, stream>>>(xh, Wqh, Wkh, Wvh,
                                                   bq, bk, bv, Qb, Kb, Vt);

  // 3) ultrametric attention (VALU distance + WMMA P@V)
  ultra_attn<<<dim3(SEQ / 16, NHEADS, BATCH), 32, 0, stream>>>(Qb, Kb, Vt, Att);

  // 4) output projection (WMMA)
  out_proj<<<dim3(SALL / 16, EMBED / 16), 32, 0, stream>>>(Att, Woh, bo, out);
}
